// PoincareSegmentEncoder_87668872446030
// MI455X (gfx1250) — matrix operand order repaired and checked by reading, not compiled, not run on hardware
//
#include <hip/hip_runtime.h>
#include <hip/hip_bf16.h>

// ---------------------------------------------------------------------------
// PoincareSegmentEncoder for MI455X (gfx1250, wave32, WMMA + TDM/async-LDS)
//   B=1024, T=336, C=128, SEG=24, NSEG=14, SEGC=3072, D=128
// ---------------------------------------------------------------------------

typedef __attribute__((ext_vector_type(16))) __bf16       v16bf;
typedef __attribute__((ext_vector_type(8)))  float        v8f;
typedef __attribute__((ext_vector_type(4)))  unsigned int u32x4;
typedef __attribute__((ext_vector_type(4)))  int          i32x4;
typedef __attribute__((ext_vector_type(8)))  int          i32x8;

#define NB    1024
#define NSEGS 14
#define SEGC  3072
#define DD    128
#define GB    2             // batches per block (shares B fragments -> halves L2 traffic)
#define KC    256           // K-chunk staged in LDS per outer step
#define KOS   12            // SEGC / KC

// Staging path selection (device pass only; host pass sees sync fallback)
#if defined(__HIP_DEVICE_COMPILE__) && defined(__has_builtin)
  #if __has_builtin(__builtin_amdgcn_tensor_load_to_lds) && \
      __has_builtin(__builtin_amdgcn_s_wait_tensorcnt)
    #define STAGE_TDM 1
  #elif __has_builtin(__builtin_amdgcn_global_load_async_to_lds_b128) && \
        __has_builtin(__builtin_amdgcn_s_wait_asynccnt)
    #define STAGE_ASYNC 1
  #endif
#endif

typedef __attribute__((address_space(3))) float       lds_f32;
typedef __attribute__((address_space(3))) void        lds_void;
typedef const __attribute__((address_space(1))) void  glb_cvoid;

// 16-bit A-matrix 16x32 per-lane element->K mapping (contiguous: k = 16*(lane/16) + e, matches B frags)
__device__ __forceinline__ int a_klocal(int lane, int e) {
    return (lane < 16) ? e : (16 + e);
}

// ---------------------------------------------------------------------------
// Prep kernels: pre-swizzle W1/W2 into CDNA5 B-operand fragment order (bf16)
// ---------------------------------------------------------------------------
__global__ __launch_bounds__(256)
void prep_w1_frags(const float* __restrict__ W1, __bf16* __restrict__ w1f) {
    int blk = blockIdx.x;                 // (s*96 + kc)*8 + nt ; 3072 blocks
    int nt = blk & 7; int skc = blk >> 3; int kc = skc % 96; int s = skc / 96;
    #pragma unroll
    for (int i = 0; i < 2; ++i) {
        int q = threadIdx.x + i * 256;    // 0..511
        int lane = q >> 4, e = q & 15;
        int K = kc * 32 + ((lane < 16) ? e : 16 + e);
        int N = nt * 16 + (lane & 15);
        w1f[(size_t)blk * 512 + q] = (__bf16)W1[((size_t)s * SEGC + K) * DD + N];
    }
}

__global__ __launch_bounds__(256)
void prep_w2_frags(const float* __restrict__ W2, __bf16* __restrict__ w2f) {
    int blk = blockIdx.x;                 // (s*4 + kc)*8 + nt ; 64 blocks
    int nt = blk & 7; int skc = blk >> 3; int kc = skc % 4; int s = skc / 4;
    #pragma unroll
    for (int i = 0; i < 2; ++i) {
        int q = threadIdx.x + i * 256;
        int lane = q >> 4, e = q & 15;
        int K = kc * 32 + ((lane < 16) ? e : 16 + e);
        int N = nt * 16 + (lane & 15);
        w2f[(size_t)blk * 512 + q] = (__bf16)W2[((size_t)s * DD + K) * DD + N];
    }
}

__global__ __launch_bounds__(256)
void prep_colsum(const float* __restrict__ W1, float* __restrict__ colsum) {
    int idx = blockIdx.x * 256 + threadIdx.x;   // 512 = 4 streams * 128 cols
    if (idx >= 4 * DD) return;
    int s = idx >> 7, n = idx & 127;
    float acc = 0.f;
    for (int k = 0; k < SEGC; ++k) acc += W1[((size_t)s * SEGC + k) * DD + n];
    colsum[idx] = acc;
}

// ---------------------------------------------------------------------------
// Encoder: one block per (stream, batch-pair). 256 threads = 8 waves.
// M=32 (2 batches x 16-row tiles) x N=128 x K=3072 ; each wave owns one
// N-tile and computes both M-tiles, reusing the B fragment.
// ---------------------------------------------------------------------------
__global__ __launch_bounds__(256)
void enc_kernel(const float* __restrict__ x0, const float* __restrict__ x1,
                const float* __restrict__ x2, const float* __restrict__ x3,
                const __bf16* __restrict__ w1f, const __bf16* __restrict__ w2f,
                const float* __restrict__ colsum,
                const float* __restrict__ b1,  const float* __restrict__ ln_g,
                const float* __restrict__ ln_b, const float* __restrict__ b2,
                const float* __restrict__ scale_p, float* __restrict__ out)
{
    __shared__ float  Xbuf[32][KC];       // 32 KB  raw f32 A chunk (2 batches)
    __shared__ float  Hbuf[32][DD];       // 16 KB  f32 intermediates
    __shared__ __bf16 A2buf[32][DD];      // 8 KB   bf16 GEMM2 A
    __shared__ float  psum[256], psq[256];
    __shared__ float  smu[32], sisd[32], slnm[32], slnr[32], sfac[32];

    const int s  = blockIdx.x >> 9;       // 4 streams x 512 pairs
    const int b0 = (blockIdx.x & 511) * GB;
    const float* xs = (s == 0 ? x0 : s == 1 ? x1 : s == 2 ? x2 : x3);
    const float* xb[GB];
    xb[0] = xs + (size_t)(b0 + 0) * (NSEGS * SEGC);
    xb[1] = xs + (size_t)(b0 + 1) * (NSEGS * SEGC);

    const int tid  = threadIdx.x;
    const int lane = tid & 31;
    const int wave = tid >> 5;
    const int lrow = tid >> 3;            // 0..31 (staging/stat row)
    const int l8   = tid & 7;
    const int g    = lrow >> 4;           // batch within block
    const int srow = lrow & 15;           // segment row within batch

    // zero padding rows 14,15 and 30,31 once (never written by staging)
    #pragma unroll
    for (int i = tid; i < 2 * KC; i += 256) {
        (&Xbuf[14][0])[i] = 0.f;
        (&Xbuf[30][0])[i] = 0.f;
    }

    float runS = 0.f, runQ = 0.f;         // per-row raw sums (threads 0..31)
    v8f acc0 = {}, acc1 = {};

    for (int ko = 0; ko < KOS; ++ko) {
        // ================= stage 2 x (14 x 256) f32 tiles into LDS =========
#if defined(STAGE_TDM)
        if (wave == 0) {
            #pragma unroll
            for (int gg = 0; gg < GB; ++gg) {
                unsigned long long ga =
                    (unsigned long long)(uintptr_t)(xb[gg] + (size_t)ko * KC);
                unsigned ldsb =
                    (unsigned)(unsigned long long)(lds_f32*)&Xbuf[gg * 16][0];
                // D# group0: count=1, lds_addr, global_addr, type=2
                u32x4 g0 = { 1u, ldsb, (unsigned)(ga & 0xFFFFFFFFu),
                             (unsigned)((ga >> 32) & 0x01FFFFFFu) | 0x80000000u };
                // D# group1: data_size=4B, tensor 3072x14, tile 256x14, stride 3072
                i32x8 g1 = { (int)(2u << 16),
                             (int)((SEGC & 0xFFFFu) << 16),
                             (int)(((unsigned)NSEGS) << 16),
                             (int)(((unsigned)KC) << 16),
                             (int)NSEGS,
                             (int)SEGC,
                             0, 0 };
                i32x4 z4 = { 0, 0, 0, 0 };
#if __clang_major__ >= 23
                i32x8 z8 = { 0, 0, 0, 0, 0, 0, 0, 0 };
                __builtin_amdgcn_tensor_load_to_lds(g0, g1, z4, z4, z8, 0);
#else
                __builtin_amdgcn_tensor_load_to_lds(g0, g1, z4, z4, 0);
#endif
            }
            __builtin_amdgcn_s_wait_tensorcnt(0);
        }
        __syncthreads();
#elif defined(STAGE_ASYNC)
        {
            const float* src = xb[g] + (size_t)srow * SEGC + ko * KC + l8 * 32;
            if (srow < NSEGS) {
                #pragma unroll
                for (int j = 0; j < 8; ++j)
                    __builtin_amdgcn_global_load_async_to_lds_b128(
                        (glb_cvoid*)(src + j * 4),
                        (lds_void*)&Xbuf[lrow][l8 * 32 + j * 4], 0, 0);
            }
            __builtin_amdgcn_s_wait_asynccnt(0);
        }
        __syncthreads();
#else
        {
            const float* src = xb[g] + (size_t)srow * SEGC + ko * KC + l8 * 32;
            if (srow < NSEGS && ko + 1 < KOS)
                __builtin_prefetch(src + KC, 0, 1);   // global_prefetch_b8
            #pragma unroll
            for (int j = 0; j < 8; ++j) {
                float4 v4 = make_float4(0.f, 0.f, 0.f, 0.f);
                if (srow < NSEGS) v4 = ((const float4*)src)[j];
                int jj = l8 * 32 + j * 4;
                Xbuf[lrow][jj + 0] = v4.x;
                Xbuf[lrow][jj + 1] = v4.y;
                Xbuf[lrow][jj + 2] = v4.z;
                Xbuf[lrow][jj + 3] = v4.w;
            }
        }
        __syncthreads();
#endif
        // ================= per-row raw stats from LDS ======================
        float ls = 0.f, lq = 0.f;
        #pragma unroll
        for (int j = 0; j < 32; ++j) {
            float v = Xbuf[lrow][l8 * 32 + j];
            ls += v; lq += v * v;
        }
        psum[tid] = ls; psq[tid] = lq;
        __syncthreads();
        if (tid < 32) {
            float a = 0.f, q = 0.f;
            #pragma unroll
            for (int i = 0; i < 8; ++i) { a += psum[tid * 8 + i]; q += psq[tid * 8 + i]; }
            runS += a; runQ += q;
        }
        // ================= WMMA: 8 k-steps, 2 M-tiles, shared B ===========
        #pragma unroll
        for (int kk = 0; kk < KC / 32; ++kk) {
            int kcg = ko * (KC / 32) + kk;
            v16bf bfrag = *(const v16bf*)(w1f + ((size_t)((s * 96 + kcg) * 8 + wave)) * 512
                                              + lane * 16);
            v16bf af0, af1;
            #pragma unroll
            for (int e = 0; e < 16; ++e) {
                int kl = kk * 32 + a_klocal(lane, e);
                af0[e] = (__bf16)Xbuf[(lane & 15)][kl];
                af1[e] = (__bf16)Xbuf[16 + (lane & 15)][kl];
            }
            acc0 = __builtin_amdgcn_wmma_f32_16x16x32_bf16(
                       false, af0, false, bfrag, (short)0, acc0, false, false);
            acc1 = __builtin_amdgcn_wmma_f32_16x16x32_bf16(
                       false, af1, false, bfrag, (short)0, acc1, false, false);
        }
        __syncthreads();
    }

    // ---- segment stats: mu, 1/(unbiased std + 1e-6) ----
    if (tid < 32) {
        float mean = runS * (1.0f / 3072.0f);
        float var  = (runQ - 3072.0f * mean * mean) * (1.0f / 3071.0f);
        float sd   = sqrtf(fmaxf(var, 0.f)) + 1e-6f;
        smu[tid] = mean; sisd[tid] = 1.0f / sd;
    }
    __syncthreads();

    // ---- GEMM1 tiles with standardization fixup + b1 ----
    {
        int N = wave * 16 + (lane & 15);
        float cs = colsum[s * DD + N];
        float bb = b1[s * DD + N];
        #pragma unroll
        for (int r = 0; r < 8; ++r) {
            int M = r + ((lane >= 16) ? 8 : 0);
            Hbuf[M][N]      = (acc0[r] - smu[M] * cs)      * sisd[M]      + bb;
            Hbuf[16 + M][N] = (acc1[r] - smu[16 + M] * cs) * sisd[16 + M] + bb;
        }
    }
    __syncthreads();

    // ---- LayerNorm(128) stats ----
    if (tid < 32) {
        float m = 0.f, q = 0.f;
        for (int j = 0; j < DD; ++j) { float h = Hbuf[tid][j]; m += h; q += h * h; }
        m *= (1.0f / DD);
        float v = q * (1.0f / DD) - m * m;
        slnm[tid] = m; slnr[tid] = rsqrtf(v + 1e-5f);
    }
    __syncthreads();

    // ---- apply LN + exact GELU -> bf16 A for GEMM2 ----
    #pragma unroll
    for (int i = 0; i < 16; ++i) {
        int e = tid + 256 * i;            // 0..4095
        int row = e >> 7, col = e & 127;
        float h = (Hbuf[row][col] - slnm[row]) * slnr[row] * ln_g[s * DD + col]
                  + ln_b[s * DD + col];
        float gel = 0.5f * h * (1.0f + erff(h * 0.7071067811865476f));
        A2buf[row][col] = (__bf16)gel;
    }
    __syncthreads();

    // ---- GEMM2: 32x128x128 (2 M-tiles per wave, shared B) ----
    v8f acc20 = {}, acc21 = {};
    #pragma unroll
    for (int kk = 0; kk < 4; ++kk) {
        v16bf bfrag = *(const v16bf*)(w2f + ((size_t)((s * 4 + kk) * 8 + wave)) * 512
                                          + lane * 16);
        v16bf af0, af1;
        #pragma unroll
        for (int e = 0; e < 16; ++e) {
            int kl = kk * 32 + a_klocal(lane, e);
            af0[e] = A2buf[(lane & 15)][kl];
            af1[e] = A2buf[16 + (lane & 15)][kl];
        }
        acc20 = __builtin_amdgcn_wmma_f32_16x16x32_bf16(
                    false, af0, false, bfrag, (short)0, acc20, false, false);
        acc21 = __builtin_amdgcn_wmma_f32_16x16x32_bf16(
                    false, af1, false, bfrag, (short)0, acc21, false, false);
    }

    // ---- v = tanh(scale) * (z + b2) -> Hbuf ----
    float ts = tanhf(scale_p[0]);
    {
        int N = wave * 16 + (lane & 15);
        float bb = b2[s * DD + N];
        #pragma unroll
        for (int r = 0; r < 8; ++r) {
            int M = r + ((lane >= 16) ? 8 : 0);
            Hbuf[M][N]      = ts * (acc20[r] + bb);
            Hbuf[16 + M][N] = ts * (acc21[r] + bb);
        }
    }
    __syncthreads();

    // ---- expmap0 + projx factor per row ----
    if (tid < 32) {
        float q = 0.f;
        for (int j = 0; j < DD; ++j) { float h = Hbuf[tid][j]; q += h * h; }
        float n  = fmaxf(sqrtf(q), 1e-15f);
        float tn = tanhf(n);
        const float maxn = 1.0f - 4e-3f;
        sfac[tid] = ((tn > maxn) ? maxn : tn) / n;
    }
    __syncthreads();

    // ---- store z_all for both batches (rows 0..13 of each tile) ----
    #pragma unroll
    for (int gg = 0; gg < GB; ++gg) {
        size_t obase = ((size_t)(s * NB + b0 + gg)) * (NSEGS * DD);
        #pragma unroll
        for (int i = 0; i < 7; ++i) {
            int e = tid + 256 * i;        // 0..1791 = 14*128
            int row = e >> 7, col = e & 127;
            out[obase + e] = sfac[gg * 16 + row] * Hbuf[gg * 16 + row][col];
        }
    }
}

// ---------------------------------------------------------------------------
// Block-wide sum over 128 threads (4 waves): shuffle tree + LDS combine
// ---------------------------------------------------------------------------
__device__ __forceinline__ float bsum(float v, float* sred) {
    #pragma unroll
    for (int off = 16; off > 0; off >>= 1) v += __shfl_down(v, off, 32);
    int lane = threadIdx.x & 31, w = threadIdx.x >> 5;
    __syncthreads();                       // protect sred reuse across calls
    if (lane == 0) sred[w] = v;
    __syncthreads();
    return sred[0] + sred[1] + sred[2] + sred[3];
}

// ---------------------------------------------------------------------------
// Frechet (Karcher) mean: one block per (stream, batch), thread = dimension
// ---------------------------------------------------------------------------
__global__ __launch_bounds__(128)
void mean_kernel(const float* __restrict__ zall, float* __restrict__ pooled)
{
    __shared__ float sred[4];
    const int s = blockIdx.x >> 10;
    const int b = blockIdx.x & 1023;
    const int c = threadIdx.x;
    const float* zb = zall + ((size_t)(s * NB + b)) * (NSEGS * DD);

    const float MINV = 1e-15f, AT_MAX = 1.0f - 1e-7f, MAXN = 1.0f - 4e-3f;

    float p[NSEGS], y2[NSEGS];
    #pragma unroll
    for (int n = 0; n < NSEGS; ++n) p[n] = zb[n * DD + c];
    #pragma unroll
    for (int n = 0; n < NSEGS; ++n) y2[n] = bsum(p[n] * p[n], sred);

    // init: projx(expmap0(mean_n logmap0(p_n)))
    float accv = 0.f;
    #pragma unroll
    for (int n = 0; n < NSEGS; ++n) {
        float nn = fmaxf(sqrtf(y2[n]), MINV);
        accv += (atanhf(fminf(nn, AT_MAX)) / nn) * p[n];
    }
    accv *= (1.0f / NSEGS);
    float u2  = bsum(accv * accv, sred);
    float nu  = fmaxf(sqrtf(u2), MINV);
    float tnu = tanhf(nu);
    float cur = (tnu / nu) * accv;
    if (tnu > MAXN) cur *= MAXN / fmaxf(tnu, MINV);

    for (int it = 0; it < 10; ++it) {
        float x2   = bsum(cur * cur, sred);
        float lam2 = fmaxf(1.0f - x2, MINV);          // 2/lambda(x)
        float grad = 0.f;
        #pragma unroll
        for (int n = 0; n < NSEGS; ++n) {
            float xy  = bsum(-cur * p[n], sred);      // <-x, y>
            float num = (1.0f + 2.0f * xy + y2[n]) * (-cur) + (1.0f - x2) * p[n];
            float den = fmaxf(1.0f + 2.0f * xy + x2 * y2[n], MINV);
            float w   = num / den;
            float w2  = bsum(w * w, sred);
            float nw  = fmaxf(sqrtf(w2), MINV);
            grad += lam2 * (atanhf(fminf(nw, AT_MAX)) / nw) * w;
        }
        grad *= (1.0f / NSEGS);
        // expmap(cur, 0.5*grad) = mobius_add(cur, tanh(n/lam2)*u/n)
        float uu  = 0.5f * grad;
        float uu2 = bsum(uu * uu, sred);
        float nuu = fmaxf(sqrtf(uu2), MINV);
        float t   = tanhf(nuu / lam2);
        float y   = (t / nuu) * uu;
        float yy2 = bsum(y * y, sred);
        float xyv = bsum(cur * y, sred);
        float num = (1.0f + 2.0f * xyv + yy2) * cur + (1.0f - x2) * y;
        float den = fmaxf(1.0f + 2.0f * xyv + x2 * yy2, MINV);
        float nw  = num / den;
        float n2  = bsum(nw * nw, sred);
        float nn  = fmaxf(sqrtf(n2), MINV);
        if (nn > MAXN) nw *= MAXN / nn;
        cur = nw;
    }
    pooled[((size_t)(s * NB + b)) * DD + c] = cur;
}

// ---------------------------------------------------------------------------
// Fusion: softmax weights, mobius scalar-mul + chained mobius_add + projx,
// broadcast to out[4][b][0..13][:]
// ---------------------------------------------------------------------------
__global__ __launch_bounds__(128)
void fuse_kernel(const float* __restrict__ pooled, const float* __restrict__ fw,
                 float* __restrict__ out)
{
    __shared__ float sred[4];
    const int b = blockIdx.x;
    const int c = threadIdx.x;
    const float MINV = 1e-15f, AT_MAX = 1.0f - 1e-7f, MAXN = 1.0f - 4e-3f;

    float f0 = fw[0], f1 = fw[1], f2 = fw[2], f3 = fw[3];
    float mx = fmaxf(fmaxf(f0, f1), fmaxf(f2, f3));
    float e0 = expf(f0 - mx), e1 = expf(f1 - mx), e2 = expf(f2 - mx), e3 = expf(f3 - mx);
    float inv = 1.0f / (e0 + e1 + e2 + e3);
    float w4[4] = { e0 * inv, e1 * inv, e2 * inv, e3 * inv };

    float comb;
    {
        float x  = pooled[((size_t)(0 * NB + b)) * DD + c];
        float x2 = bsum(x * x, sred);
        float n  = fmaxf(sqrtf(x2), MINV);
        comb = (tanhf(w4[0] * atanhf(fminf(n, AT_MAX))) / n) * x;
    }
    for (int i = 1; i < 4; ++i) {
        float x  = pooled[((size_t)(i * NB + b)) * DD + c];
        float x2 = bsum(x * x, sred);
        float n  = fmaxf(sqrtf(x2), MINV);
        float q  = (tanhf(w4[i] * atanhf(fminf(n, AT_MAX))) / n) * x;
        float c2 = bsum(comb * comb, sred);
        float q2 = bsum(q * q, sred);
        float cq = bsum(comb * q, sred);
        float num = (1.0f + 2.0f * cq + q2) * comb + (1.0f - c2) * q;
        float den = fmaxf(1.0f + 2.0f * cq + c2 * q2, MINV);
        comb = num / den;
    }
    float c2 = bsum(comb * comb, sred);
    float n  = fmaxf(sqrtf(c2), MINV);
    if (n > MAXN) comb *= MAXN / n;

    size_t obase = ((size_t)(4 * NB + b)) * (NSEGS * DD);
    #pragma unroll
    for (int nseg = 0; nseg < NSEGS; ++nseg)
        out[obase + nseg * DD + c] = comb;
}

// ---------------------------------------------------------------------------
extern "C" void kernel_launch(void* const* d_in, const int* in_sizes, int n_in,
                              void* d_out, int out_size, void* d_ws, size_t ws_size,
                              hipStream_t stream) {
    (void)in_sizes; (void)n_in; (void)out_size; (void)ws_size;
    const float* x0   = (const float*)d_in[0];
    const float* x1   = (const float*)d_in[1];
    const float* x2   = (const float*)d_in[2];
    const float* x3   = (const float*)d_in[3];
    const float* W1   = (const float*)d_in[4];
    const float* b1   = (const float*)d_in[5];
    const float* ln_g = (const float*)d_in[6];
    const float* ln_b = (const float*)d_in[7];
    const float* W2   = (const float*)d_in[8];
    const float* b2   = (const float*)d_in[9];
    const float* scal = (const float*)d_in[10];
    const float* fw   = (const float*)d_in[11];
    float* out = (float*)d_out;

    char* ws = (char*)d_ws;
    __bf16* w1f    = (__bf16*)ws;                            // 3,145,728 B
    __bf16* w2f    = (__bf16*)(ws + 3145728);                //   131,072 B
    float*  colsum = (float*)(ws + 3145728 + 131072);        //     2,048 B
    float*  pooled = (float*)(ws + 3145728 + 131072 + 2048); // 2,097,152 B

    prep_w1_frags<<<3072, 256, 0, stream>>>(W1, w1f);
    prep_w2_frags<<<64,   256, 0, stream>>>(W2, w2f);
    prep_colsum  <<<2,    256, 0, stream>>>(W1, colsum);
    enc_kernel   <<<2048, 256, 0, stream>>>(x0, x1, x2, x3, w1f, w2f, colsum,
                                            b1, ln_g, ln_b, b2, scal, out);
    mean_kernel  <<<4096, 128, 0, stream>>>(out, pooled);
    fuse_kernel  <<<1024, 128, 0, stream>>>(pooled, fw, out);
}
